// GraphEncoder_54185307406769
// MI455X (gfx1250) — compile-verified, matrix operand order checked by reading, not run on hardware
//
#include <hip/hip_runtime.h>
#include <stdint.h>

typedef __attribute__((ext_vector_type(16))) _Float16 v16h;
typedef __attribute__((ext_vector_type(8)))  float    v8f;

#define NN 50000
#define NE 800000
#define HID 64
#define LAT 32
#define NLAY 4
#define TILE_NODES 128

// ---------------------------------------------------------------------------
// gfx1250 async global->LDS copy (ASYNCcnt-tracked), per CDNA5 ISA 15.18.3.
// VDST = 32-bit LDS byte offset, VADDR = 64-bit global address (GV mode).
// ---------------------------------------------------------------------------
__device__ inline void async_load_b128(uint32_t ldsOff, const void* gaddr)
{
    asm volatile("global_load_async_to_lds_b128 %0, %1, off"
                 :: "v"(ldsOff), "v"((uint64_t)(uintptr_t)gaddr)
                 : "memory");
}
__device__ inline void wait_async0()
{
    asm volatile("s_wait_asynccnt 0" ::: "memory");
}
__device__ inline uint32_t lds_off(const void* p)
{
    return (uint32_t)(uintptr_t)p;   // low 32 bits of generic LDS addr = WG-relative offset
}

// ---------------------------------------------------------------------------
// Weight convert + transpose: Wt[n][k] = (f16)W[k][n]
// ---------------------------------------------------------------------------
__global__ __launch_bounds__(256) void convert_weights_kernel(
    const float* __restrict__ W1, const float* __restrict__ W2,
    const float* __restrict__ Wr,
    _Float16* __restrict__ W1t, _Float16* __restrict__ W2t,
    _Float16* __restrict__ Wrt)
{
    int idx = blockIdx.x * blockDim.x + threadIdx.x;
    const int perW = NLAY * HID * HID;            // 16384
    if (idx < perW) {
        int k = idx >> 12;
        int r = idx & 4095;
        int i = r >> 6;                           // in  (row of W)
        int j = r & 63;                           // out (col of W)
        W1t[(k << 12) + (j << 6) + i] = (_Float16)W1[idx];
        W2t[(k << 12) + (j << 6) + i] = (_Float16)W2[idx];
    } else if (idx < perW + HID * LAT) {
        int r = idx - perW;                       // Wr is [64][32]
        int i = r >> 5;
        int j = r & 31;
        Wrt[(j << 6) + i] = (_Float16)Wr[r];
    }
}

// ---------------------------------------------------------------------------
// x (f32) -> h (f16)
// ---------------------------------------------------------------------------
__global__ __launch_bounds__(256) void f32_to_f16_kernel(
    _Float16* __restrict__ dst, const float* __restrict__ src, int n4)
{
    int i = blockIdx.x * blockDim.x + threadIdx.x;
    if (i < n4) {
        float4 v = ((const float4*)src)[i];
        _Float16* d = dst + (size_t)i * 4;
        d[0] = (_Float16)v.x; d[1] = (_Float16)v.y;
        d[2] = (_Float16)v.z; d[3] = (_Float16)v.w;
    }
}

// ---------------------------------------------------------------------------
// agg (f32) = h (f16)   -- self term of GIN aggregation
// ---------------------------------------------------------------------------
__global__ __launch_bounds__(256) void f16_to_f32_kernel(
    float* __restrict__ dst, const _Float16* __restrict__ src, int n4)
{
    int i = blockIdx.x * blockDim.x + threadIdx.x;
    if (i < n4) {
        const _Float16* s = src + (size_t)i * 4;
        float4 v = make_float4((float)s[0], (float)s[1], (float)s[2], (float)s[3]);
        ((float4*)dst)[i] = v;
    }
}

// ---------------------------------------------------------------------------
// Edge scatter-add: agg[dst] += h[src].  16 threads/edge, 4 dims each.
// f16 gather (halves HBM traffic), f32 atomic accumulation (full precision).
// ---------------------------------------------------------------------------
__global__ __launch_bounds__(256) void scatter_add_kernel(
    float* __restrict__ agg, const _Float16* __restrict__ h,
    const int* __restrict__ ei)
{
    int t = blockIdx.x * blockDim.x + threadIdx.x;
    int e  = t >> 4;
    int d4 = (t & 15) << 2;
    if (e >= NE) return;
    int src = ei[e];
    int dst = ei[NE + e];
    const _Float16* hp = h + (size_t)src * HID + d4;   // 8B load
    float* a = agg + (size_t)dst * HID + d4;
    atomicAdd(a + 0, (float)hp[0]);
    atomicAdd(a + 1, (float)hp[1]);
    atomicAdd(a + 2, (float)hp[2]);
    atomicAdd(a + 3, (float)hp[3]);
}

// ---------------------------------------------------------------------------
// WMMA fragment helper (CDNA5 ISA 7.12.2, wave32).
// A 16x32 f16: lane&15 = row M; lane>>4 selects K{0..7,16..23} vs {8..15,24..31}
// Transposed weights give B the identical per-lane pattern with lane&15 = N.
// ---------------------------------------------------------------------------
__device__ inline v16h load_frag16(const _Float16* rowPtr, int kOff, int kb)
{
    v16h a;
    const _Float16* p0 = rowPtr + kOff + kb;       // 16B aligned
    const _Float16* p1 = p0 + 16;
#pragma unroll
    for (int j = 0; j < 8; ++j) { a[j] = p0[j]; a[j + 8] = p1[j]; }
    return a;
}

// ---------------------------------------------------------------------------
// GIN MLP:  h_out = (f16) [ ReLU(agg @ W1 + b1) @ W2 + b2 ]  per 128-node tile
// ---------------------------------------------------------------------------
__global__ __launch_bounds__(256) void gin_mlp_kernel(
    const float* __restrict__ agg, _Float16* __restrict__ hout,
    const _Float16* __restrict__ W1t, const float* __restrict__ b1,
    const _Float16* __restrict__ W2t, const float* __restrict__ b2)
{
    __shared__ _Float16 sTile[TILE_NODES * HID];   // 16 KB (agg tile, then ReLU tmp)
    __shared__ _Float16 sW1[HID * HID];            // 8 KB
    __shared__ _Float16 sW2[HID * HID];            // 8 KB
    __shared__ float sB1[HID];
    __shared__ float sB2[HID];

    const int tid = threadIdx.x;
    const int blockBase = blockIdx.x * TILE_NODES;

    // stage weights via async global->LDS DMA (raw f16 bytes, 512 x 16B each)
    {
        uint32_t lw1 = lds_off(sW1);
        uint32_t lw2 = lds_off(sW2);
        const char* g1 = (const char*)W1t;
        const char* g2 = (const char*)W2t;
        for (int i = tid; i < 512; i += 256) {
            async_load_b128(lw1 + i * 16, g1 + i * 16);
            async_load_b128(lw2 + i * 16, g2 + i * 16);
        }
        if (tid < HID) { sB1[tid] = b1[tid]; sB2[tid] = b2[tid]; }
    }

    // stage agg tile as f16 (f32->f16 conversion; zero-pad OOB rows)
    for (int i = tid; i < TILE_NODES * HID / 4; i += 256) {
        int e = i << 2;
        int node = blockBase + (e >> 6);
        float4 v = make_float4(0.f, 0.f, 0.f, 0.f);
        if (node < NN) v = *(const float4*)(agg + (size_t)node * HID + (e & 63));
        sTile[e + 0] = (_Float16)v.x; sTile[e + 1] = (_Float16)v.y;
        sTile[e + 2] = (_Float16)v.z; sTile[e + 3] = (_Float16)v.w;
    }
    wait_async0();
    __syncthreads();

    const int wave    = tid >> 5;
    const int lane    = tid & 31;
    const int halfSel = lane >> 4;
    const int kb      = halfSel << 3;       // K sub-offset 0 / 8
    const int col     = lane & 15;          // N within tile / C-D column
    const int rowOff  = halfSel << 3;       // C/D row offset 0 / 8

    _Float16* myTile = sTile + wave * 16 * HID;     // this wave's 16 rows
    const _Float16* aRow = myTile + (lane & 15) * HID;

    // -------- GEMM1: t = ReLU(agg @ W1 + b1) --------
    v16h a0 = load_frag16(aRow, 0, kb);
    v16h a1 = load_frag16(aRow, 32, kb);

    v8f acc[4];
#pragma unroll
    for (int nt = 0; nt < 4; ++nt) {
        int n = nt * 16 + col;
        v16h bb0 = load_frag16(sW1 + n * HID, 0, kb);
        v16h bb1 = load_frag16(sW1 + n * HID, 32, kb);
        float bv = sB1[n];
        v8f c;
#pragma unroll
        for (int v = 0; v < 8; ++v) c[v] = bv;
        c = __builtin_amdgcn_wmma_f32_16x16x32_f16(false, a0, false, bb0, (short)0, c, false, false);
        c = __builtin_amdgcn_wmma_f32_16x16x32_f16(false, a1, false, bb1, (short)0, c, false, false);
        acc[nt] = c;
    }

    // ReLU -> back to own rows of sTile (f16). Wave-private rows: no barrier.
#pragma unroll
    for (int nt = 0; nt < 4; ++nt)
#pragma unroll
        for (int v = 0; v < 8; ++v) {
            float x = acc[nt][v];
            myTile[(rowOff + v) * HID + nt * 16 + col] = (_Float16)(x > 0.f ? x : 0.f);
        }

    // -------- GEMM2: h = t @ W2 + b2 --------
    v16h t0 = load_frag16(aRow, 0, kb);
    v16h t1 = load_frag16(aRow, 32, kb);
#pragma unroll
    for (int nt = 0; nt < 4; ++nt) {
        int n = nt * 16 + col;
        v16h bb0 = load_frag16(sW2 + n * HID, 0, kb);
        v16h bb1 = load_frag16(sW2 + n * HID, 32, kb);
        float bv = sB2[n];
        v8f c;
#pragma unroll
        for (int v = 0; v < 8; ++v) c[v] = bv;
        c = __builtin_amdgcn_wmma_f32_16x16x32_f16(false, t0, false, bb0, (short)0, c, false, false);
        c = __builtin_amdgcn_wmma_f32_16x16x32_f16(false, t1, false, bb1, (short)0, c, false, false);
#pragma unroll
        for (int v = 0; v < 8; ++v) {
            int node = blockBase + wave * 16 + rowOff + v;
            if (node < NN) hout[(size_t)node * HID + n] = (_Float16)c[v];
        }
    }
}

// ---------------------------------------------------------------------------
// Final: z = h @ Wr + br ; out = z / max(||z||, 1e-12)
// ---------------------------------------------------------------------------
__global__ __launch_bounds__(256) void final_proj_kernel(
    const _Float16* __restrict__ h, const _Float16* __restrict__ Wrt,
    const float* __restrict__ br, float* __restrict__ out)
{
    __shared__ _Float16 sTile[TILE_NODES * HID];   // 16 KB
    __shared__ _Float16 sWr[LAT * HID];            // 4 KB
    __shared__ float    sBr[LAT];
    __shared__ float    zTile[TILE_NODES * LAT];   // 16 KB

    const int tid = threadIdx.x;
    const int blockBase = blockIdx.x * TILE_NODES;

    // weights: 256 x 16B async chunks
    {
        uint32_t lw = lds_off(sWr);
        const char* gw = (const char*)Wrt;
        for (int i = tid; i < 256; i += 256)
            async_load_b128(lw + i * 16, gw + i * 16);
        if (tid < LAT) sBr[tid] = br[tid];
    }
    // h tile: raw f16 copy, 1024 x 16B chunks (8 halves each; row = i>>3)
    {
        uint32_t lt = lds_off(sTile);
        const char* gh = (const char*)(h + (size_t)blockBase * HID);
        for (int i = tid; i < 1024; i += 256) {
            int node = blockBase + (i >> 3);
            if (node < NN) {
                async_load_b128(lt + i * 16, gh + i * 16);
            } else {
                uint4 z = make_uint4(0u, 0u, 0u, 0u);
                ((uint4*)sTile)[i] = z;
            }
        }
    }
    wait_async0();
    __syncthreads();

    const int wave    = tid >> 5;
    const int lane    = tid & 31;
    const int halfSel = lane >> 4;
    const int kb      = halfSel << 3;
    const int col     = lane & 15;
    const int rowOff  = halfSel << 3;

    const _Float16* aRow = sTile + (wave * 16 + (lane & 15)) * HID;
    v16h a0 = load_frag16(aRow, 0, kb);
    v16h a1 = load_frag16(aRow, 32, kb);

#pragma unroll
    for (int nt = 0; nt < 2; ++nt) {
        int n = nt * 16 + col;
        v16h bb0 = load_frag16(sWr + n * HID, 0, kb);
        v16h bb1 = load_frag16(sWr + n * HID, 32, kb);
        float bv = sBr[n];
        v8f c;
#pragma unroll
        for (int v = 0; v < 8; ++v) c[v] = bv;
        c = __builtin_amdgcn_wmma_f32_16x16x32_f16(false, a0, false, bb0, (short)0, c, false, false);
        c = __builtin_amdgcn_wmma_f32_16x16x32_f16(false, a1, false, bb1, (short)0, c, false, false);
#pragma unroll
        for (int v = 0; v < 8; ++v)
            zTile[(wave * 16 + rowOff + v) * LAT + n] = c[v];
    }
    __syncthreads();

    if (tid < TILE_NODES) {
        int node = blockBase + tid;
        if (node < NN) {
            float s = 0.f;
#pragma unroll
            for (int j = 0; j < LAT; ++j) { float z = zTile[tid * LAT + j]; s += z * z; }
            float inv = 1.0f / fmaxf(sqrtf(s), 1e-12f);
#pragma unroll
            for (int j = 0; j < LAT; ++j)
                out[(size_t)node * LAT + j] = zTile[tid * LAT + j] * inv;
        }
    }
}

// ---------------------------------------------------------------------------
extern "C" void kernel_launch(void* const* d_in, const int* in_sizes, int n_in,
                              void* d_out, int out_size, void* d_ws, size_t ws_size,
                              hipStream_t stream)
{
    (void)in_sizes; (void)n_in; (void)out_size; (void)ws_size;

    const float* x  = (const float*)d_in[0];
    const int*   ei = (const int*)d_in[1];
    const float* W1 = (const float*)d_in[2];
    const float* b1 = (const float*)d_in[3];
    const float* W2 = (const float*)d_in[4];
    const float* b2 = (const float*)d_in[5];
    const float* Wr = (const float*)d_in[6];
    const float* br = (const float*)d_in[7];
    float* out = (float*)d_out;

    // workspace: agg f32 (12.8MB) | h f16 (6.4MB) | W1t,W2t,Wrt f16
    float*    agg = (float*)d_ws;
    _Float16* h   = (_Float16*)(agg + (size_t)NN * HID);
    _Float16* W1t = h + (size_t)NN * HID;
    _Float16* W2t = W1t + NLAY * HID * HID;
    _Float16* Wrt = W2t + NLAY * HID * HID;

    const int nConv = NLAY * HID * HID + HID * LAT;              // 18432
    convert_weights_kernel<<<(nConv + 255) / 256, 256, 0, stream>>>(
        W1, W2, Wr, W1t, W2t, Wrt);

    const int n4 = NN * HID / 4;                                 // 800000
    f32_to_f16_kernel<<<(n4 + 255) / 256, 256, 0, stream>>>(h, x, n4);

    const int mlpBlocks = (NN + TILE_NODES - 1) / TILE_NODES;    // 391
    const int scatThreads = NE * 16;                             // 12.8M
    for (int k = 0; k < NLAY; ++k) {
        f16_to_f32_kernel<<<(n4 + 255) / 256, 256, 0, stream>>>(agg, h, n4);
        scatter_add_kernel<<<(scatThreads + 255) / 256, 256, 0, stream>>>(agg, h, ei);
        gin_mlp_kernel<<<mlpBlocks, 256, 0, stream>>>(
            agg, h,
            W1t + k * HID * HID, b1 + k * HID,
            W2t + k * HID * HID, b2 + k * HID);
    }

    final_proj_kernel<<<mlpBlocks, 256, 0, stream>>>(h, Wrt, br, out);
}
